// alinet_model_77163382440892
// MI455X (gfx1250) — matrix operand order repaired and checked by reading, not compiled, not run on hardware
//
#include <hip/hip_runtime.h>
#include <hip/hip_bf16.h>

typedef __attribute__((ext_vector_type(16))) _Float16 v16h;
typedef __attribute__((ext_vector_type(8)))  _Float16 v8h;
typedef __attribute__((ext_vector_type(8)))  float    v8f;

#define DD 128
#define BN_SCALE 0.9995003746877732f
#define ENC_NEG_INF 0x007FFFFFu  // monotonic-uint encoding of -inf

// ---------------------------------------------------------------------------
// Weight pre-swizzle: f32 row-major (K=128 x N=128) -> f16 WMMA-B fragments.
// Slot layout: [mat][ct(8)][kc(4)][lane(32)][16 f16]; element e of a slot is
// W[k = kc*32 + (lane<16?0:16) + e][n = ct*16 + (lane&15)].
// ---------------------------------------------------------------------------
__global__ __launch_bounds__(256) void prep_weights(
    const float* __restrict__ w0, const float* __restrict__ w1,
    const float* __restrict__ w2, const float* __restrict__ w3,
    const float* __restrict__ w4, const float* __restrict__ w5,
    _Float16* __restrict__ wswz) {
  int tid = blockIdx.x * 256 + threadIdx.x;      // 6*8*4*32 = 6144 slots
  if (tid >= 6 * 8 * 4 * 32) return;
  int lane = tid & 31;
  int kc   = (tid >> 5) & 3;
  int ct   = (tid >> 7) & 7;
  int m    = tid >> 10;
  const float* W = (m == 0) ? w0 : (m == 1) ? w1 : (m == 2) ? w2
                 : (m == 3) ? w3 : (m == 4) ? w4 : w5;
  _Float16* dst = wswz + (size_t)tid * 16;
  int kbase = kc * 32 + ((lane < 16) ? 0 : 16);
  int n = ct * 16 + (lane & 15);
#pragma unroll
  for (int e = 0; e < 16; ++e)
    dst[e] = (_Float16)W[(kbase + e) * DD + n];
}

// ---------------------------------------------------------------------------
// C[M x 128] = (scale * A[M x 128]) @ W  via v_wmma_f32_16x16x32_f16.
// Block = 256 threads (8 waves). Block tile = 64 rows x 128 cols:
// each wave owns one 16-col tile and 4 row sub-tiles (4 accumulators),
// so each B fragment is reused by 4 WMMAs. 16 WMMAs per wave total.
// ---------------------------------------------------------------------------
__global__ __launch_bounds__(256) void gemm_bn_wmma(
    const float* __restrict__ A, const _Float16* __restrict__ Bswz,
    float* __restrict__ C, int M, float scale) {
  __shared__ _Float16 As[64 * 136];  // 136-f16 stride: 16B-aligned, bank-rotated
  const int row0 = blockIdx.x * 64;
  const int t = threadIdx.x;
  const bool full = (row0 + 64 <= M);
  if (full) {
#pragma unroll
    for (int i = 0; i < 32; ++i) {
      int idx = t + i * 256;          // 64*128 = 8192 elements
      int r = idx >> 7, c = idx & 127;
      As[r * 136 + c] = (_Float16)(A[(size_t)(row0 + r) * DD + c] * scale);
    }
  } else {
#pragma unroll
    for (int i = 0; i < 32; ++i) {
      int idx = t + i * 256;
      int r = idx >> 7, c = idx & 127;
      float v = (row0 + r < M) ? A[(size_t)(row0 + r) * DD + c] : 0.0f;
      As[r * 136 + c] = (_Float16)(v * scale);
    }
  }
  __syncthreads();

  const int wave = t >> 5, lane = t & 31;
  const int ct = wave;                       // 8 waves -> 8 column tiles
  const int r0 = lane & 15;
  const int halfsel = (lane < 16) ? 0 : 16;
  v8f acc[4] = {};
#pragma unroll
  for (int kc = 0; kc < 4; ++kc) {
    const int koff = kc * 32 + halfsel;
    union { v16h v; v8h h[2]; } b;
    const _Float16* bp = Bswz + ((size_t)((ct * 4 + kc) * 32 + lane)) * 16;
    b.h[0] = *(const v8h*)(bp);
    b.h[1] = *(const v8h*)(bp + 8);
#pragma unroll
    for (int rt = 0; rt < 4; ++rt) {
      union { v16h v; v8h h[2]; } a;
      const _Float16* ap = &As[(rt * 16 + r0) * 136 + koff];
      a.h[0] = *(const v8h*)(ap);
      a.h[1] = *(const v8h*)(ap + 8);
      acc[rt] = __builtin_amdgcn_wmma_f32_16x16x32_f16(
          false, a.v, false, b.v, (short)0, acc[rt], false, false);
    }
  }
  const int n = ct * 16 + (lane & 15);
  const int mloc = (lane < 16) ? 0 : 8;
  if (full) {
#pragma unroll
    for (int rt = 0; rt < 4; ++rt) {
      float* crow = C + (size_t)(row0 + rt * 16 + mloc) * DD + n;
#pragma unroll
      for (int j = 0; j < 8; ++j) crow[(size_t)j * DD] = acc[rt][j];
    }
  } else {
#pragma unroll
    for (int rt = 0; rt < 4; ++rt) {
#pragma unroll
      for (int j = 0; j < 8; ++j) {
        int m = row0 + rt * 16 + mloc + j;
        if (m < M) C[(size_t)m * DD + n] = acc[rt][j];
      }
    }
  }
}

// ---------------------------------------------------------------------------
// s1/s2: per-row tanh( sum_d p*(x*BN) ). One wave per row.
// ---------------------------------------------------------------------------
__global__ __launch_bounds__(256) void s12_kernel(
    const float* __restrict__ x, const float* __restrict__ p1,
    const float* __restrict__ p2, float* __restrict__ s1,
    float* __restrict__ s2, int N) {
  int row = blockIdx.x * 8 + (threadIdx.x >> 5);
  int lane = threadIdx.x & 31;
  if (row >= N) return;
  size_t base = (size_t)row * DD;
  float a1 = 0.f, a2 = 0.f;
#pragma unroll
  for (int j = 0; j < 4; ++j) {
    int d = lane + j * 32;
    float xb = x[base + d] * BN_SCALE;
    a1 += p1[base + d] * xb;
    a2 += p2[base + d] * xb;
  }
#pragma unroll
  for (int off = 16; off; off >>= 1) {
    a1 += __shfl_xor(a1, off, 32);
    a2 += __shfl_xor(a2, off, 32);
  }
  if (lane == 0) { s1[row] = tanhf(a1); s2[row] = tanhf(a2); }
}

__global__ void fill_f32(float* __restrict__ p, float v, int n) {
  int i = blockIdx.x * 256 + threadIdx.x;
  if (i < n) p[i] = v;
}
__global__ void fill_u32(unsigned* __restrict__ p, unsigned v, int n) {
  int i = blockIdx.x * 256 + threadIdx.x;
  if (i < n) p[i] = v;
}

// ---------------------------------------------------------------------------
// SpMM scatter: out[r,:] += v * h[c,:]. One wave per edge, 4 floats per lane.
// ---------------------------------------------------------------------------
__global__ __launch_bounds__(256) void spmm_scatter(
    const int* __restrict__ rows, const int* __restrict__ cols,
    const float* __restrict__ vals, const float* __restrict__ h,
    float* __restrict__ out, int E) {
  long long gid = (long long)blockIdx.x * 256 + threadIdx.x;
  int edge = (int)(gid >> 5);
  int lane = (int)(gid & 31);
  if (edge >= E) return;
  if (lane == 0 && edge + 512 < E) {   // stream-ahead on index arrays
    __builtin_prefetch(&rows[edge + 512], 0, 0);
    __builtin_prefetch(&cols[edge + 512], 0, 0);
    __builtin_prefetch(&vals[edge + 512], 0, 0);
  }
  int r = rows[edge], c = cols[edge];
  float v = vals[edge];
  const float* hc = h + (size_t)c * DD;
  float* orow = out + (size_t)r * DD;
#pragma unroll
  for (int j = 0; j < 4; ++j) {
    int d = lane + j * 32;
    atomicAdd(&orow[d], v * hc[d]);
  }
}

// Attention-weighted scatter: out[r,:] += (ex/denom[r]) * mapped[c,:]
__global__ __launch_bounds__(256) void att_scatter(
    const int* __restrict__ rows, const int* __restrict__ cols,
    const float* __restrict__ ex, const float* __restrict__ denom,
    const float* __restrict__ mapped, float* __restrict__ out, int E) {
  long long gid = (long long)blockIdx.x * 256 + threadIdx.x;
  int edge = (int)(gid >> 5);
  int lane = (int)(gid & 31);
  if (edge >= E) return;
  if (lane == 0 && edge + 512 < E) {
    __builtin_prefetch(&rows[edge + 512], 0, 0);
    __builtin_prefetch(&cols[edge + 512], 0, 0);
    __builtin_prefetch(&ex[edge + 512], 0, 0);
  }
  int r = rows[edge], c = cols[edge];
  float a = ex[edge] / denom[r];
  const float* mc = mapped + (size_t)c * DD;
  float* orow = out + (size_t)r * DD;
#pragma unroll
  for (int j = 0; j < 4; ++j) {
    int d = lane + j * 32;
    atomicAdd(&orow[d], a * mc[d]);
  }
}

// Edge scores: e = leaky_relu(v*s1[r] + v*s2[c], 0.2); segment max via
// monotonic-uint atomicMax.
__global__ void edge_score(
    const int* __restrict__ rows, const int* __restrict__ cols,
    const float* __restrict__ vals, const float* __restrict__ s1,
    const float* __restrict__ s2, float* __restrict__ e,
    unsigned* __restrict__ mxu, int E) {
  int i = blockIdx.x * 256 + threadIdx.x;
  if (i >= E) return;
  float v = vals[i];
  float t = v * s1[rows[i]] + v * s2[cols[i]];
  float lr = (t > 0.f) ? t : 0.2f * t;
  e[i] = lr;
  unsigned b = __float_as_uint(lr);
  unsigned enc = (b & 0x80000000u) ? ~b : (b | 0x80000000u);
  atomicMax(&mxu[rows[i]], enc);
}

// ex = exp(e - mx[r]); denom[r] += ex (overwrites e in place).
__global__ void edge_exp(
    const int* __restrict__ rows, float* __restrict__ e,
    const unsigned* __restrict__ mxu, float* __restrict__ denom, int E) {
  int i = blockIdx.x * 256 + threadIdx.x;
  if (i >= E) return;
  unsigned enc = mxu[rows[i]];
  unsigned b = (enc & 0x80000000u) ? (enc & 0x7FFFFFFFu) : ~enc;
  float mx = __uint_as_float(b);
  float ex = expf(e[i] - mx);
  e[i] = ex;
  atomicAdd(&denom[rows[i]], ex);
}

// z = tanh(z + bias[d]) in place
__global__ void bias_tanh(float* __restrict__ z, const float* __restrict__ b, int n) {
  int i = blockIdx.x * 256 + threadIdx.x;
  if (i < n) z[i] = tanhf(z[i] + b[i & (DD - 1)]);
}

// y3 = tanh(bn(y1)*(1-gate) + bn(y2)*gate), gate = relu(tanh(g))
__global__ void highway(
    const float* __restrict__ y1, const float* __restrict__ y2,
    const float* __restrict__ g, float* __restrict__ y3, int n) {
  int i = blockIdx.x * 256 + threadIdx.x;
  if (i >= n) return;
  float gt = tanhf(g[i]);
  float gate = (gt > 0.f) ? gt : 0.f;
  float i1 = y2[i] * BN_SCALE;  // bn(y_att)
  float i2 = y1[i] * BN_SCALE;  // bn(y_gcn)
  y3[i] = tanhf(i2 * (1.f - gate) + i1 * gate);
}

// Final: out = l2norm(concat(l2norm(y3), l2norm(y4), l2norm(x))). Wave/row.
__global__ __launch_bounds__(256) void final_norm(
    const float* __restrict__ y3, const float* __restrict__ y4,
    const float* __restrict__ x, float* __restrict__ out, int N) {
  int row = blockIdx.x * 8 + (threadIdx.x >> 5);
  int lane = threadIdx.x & 31;
  if (row >= N) return;
  size_t base = (size_t)row * DD;
  float v3[4], v4[4], vx[4];
  float s3 = 0.f, s4 = 0.f, sx = 0.f;
#pragma unroll
  for (int j = 0; j < 4; ++j) {
    int d = lane + j * 32;
    v3[j] = y3[base + d]; v4[j] = y4[base + d]; vx[j] = x[base + d];
    s3 += v3[j] * v3[j];  s4 += v4[j] * v4[j];  sx += vx[j] * vx[j];
  }
#pragma unroll
  for (int off = 16; off; off >>= 1) {
    s3 += __shfl_xor(s3, off, 32);
    s4 += __shfl_xor(s4, off, 32);
    sx += __shfl_xor(sx, off, 32);
  }
  float i3 = rsqrtf(fmaxf(s3, 1e-12f));
  float i4 = rsqrtf(fmaxf(s4, 1e-12f));
  float ix = rsqrtf(fmaxf(sx, 1e-12f));
  float so = s3 * i3 * i3 + s4 * i4 * i4 + sx * ix * ix;
  float io = rsqrtf(fmaxf(so, 1e-12f));
  float* orow = out + (size_t)row * (3 * DD);
#pragma unroll
  for (int j = 0; j < 4; ++j) {
    int d = lane + j * 32;
    orow[d]            = v3[j] * i3 * io;
    orow[DD + d]       = v4[j] * i4 * io;
    orow[2 * DD + d]   = vx[j] * ix * io;
  }
}

// ---------------------------------------------------------------------------
static inline size_t align_up(size_t v, size_t a) { return (v + a - 1) & ~(a - 1); }

extern "C" void kernel_launch(void* const* d_in, const int* in_sizes, int n_in,
                              void* d_out, int out_size, void* d_ws, size_t ws_size,
                              hipStream_t stream) {
  const float* x      = (const float*)d_in[0];
  const int*   a1r    = (const int*)  d_in[1];
  const int*   a1c    = (const int*)  d_in[2];
  const float* a1v    = (const float*)d_in[3];
  const int*   a2r    = (const int*)  d_in[4];
  const int*   a2c    = (const int*)  d_in[5];
  const float* a2v    = (const float*)d_in[6];
  const float* w_gcn1 = (const float*)d_in[7];
  const float* b_gcn1 = (const float*)d_in[8];
  const float* w_att  = (const float*)d_in[9];
  const float* m1     = (const float*)d_in[10];
  const float* m2     = (const float*)d_in[11];
  const float* w_hw   = (const float*)d_in[12];
  const float* w_gcn2 = (const float*)d_in[13];
  const float* b_gcn2 = (const float*)d_in[14];

  const int N  = in_sizes[0] / DD;
  const int E1 = in_sizes[1];
  const int E2 = in_sizes[4];
  const size_t nd = (size_t)N * DD;

  // ---- workspace carve-up -------------------------------------------------
  char* ws = (char*)d_ws;
  size_t off = 0;
  _Float16* wswz = (_Float16*)(ws + off); off = align_up(off + 6 * 128 * 128 * sizeof(_Float16), 256);
  float* bufA = (float*)(ws + off); off = align_up(off + nd * 4, 256);  // h1 -> y3
  float* bufB = (float*)(ws + off); off = align_up(off + nd * 4, 256);  // mapped -> g2
  float* bufC = (float*)(ws + off); off = align_up(off + nd * 4, 256);  // p1 -> y2
  float* bufD = (float*)(ws + off); off = align_up(off + nd * 4, 256);  // p2 -> gate -> y4
  float* bufG = (float*)(ws + off); off = align_up(off + nd * 4, 256);  // y1
  float* edgeE = (float*)(ws + off); off = align_up(off + (size_t)E2 * 4, 256);
  float* s1   = (float*)(ws + off); off = align_up(off + (size_t)N * 4, 256);
  float* s2   = (float*)(ws + off); off = align_up(off + (size_t)N * 4, 256);
  unsigned* mxu = (unsigned*)(ws + off); off = align_up(off + (size_t)N * 4, 256);
  float* denom = (float*)(ws + off); off = align_up(off + (size_t)N * 4, 256);
  (void)ws_size;

  const int gemmBlocks = (N + 63) / 64;
  const int ndBlocks   = (int)((nd + 255) / 256);
  const int rowBlocks  = (N + 7) / 8;
  const int nBlocks    = (N + 255) / 256;
  const int e1Blocks   = (E1 + 255) / 256;
  const int e2Blocks   = (E2 + 255) / 256;
  const int e1Waves    = (int)(((long long)E1 * 32 + 255) / 256);
  const int e2Waves    = (int)(((long long)E2 * 32 + 255) / 256);

  // 0. swizzle weights for WMMA B fragments
  prep_weights<<<24, 256, 0, stream>>>(w_gcn1, w_att, m1, m2, w_hw, w_gcn2, wswz);
  _Float16* WZ0 = wswz;               // w_gcn1
  _Float16* WZ1 = wswz + 16384;       // w_att
  _Float16* WZ2 = wswz + 2 * 16384;   // m1
  _Float16* WZ3 = wswz + 3 * 16384;   // m2
  _Float16* WZ4 = wswz + 4 * 16384;   // w_hw
  _Float16* WZ5 = wswz + 5 * 16384;   // w_gcn2

  // 1. stage-1 GEMMs: bn(x) @ {w_gcn1, w_att, m1, m2}
  gemm_bn_wmma<<<gemmBlocks, 256, 0, stream>>>(x, WZ0, bufA, N, BN_SCALE); // h1
  gemm_bn_wmma<<<gemmBlocks, 256, 0, stream>>>(x, WZ1, bufB, N, BN_SCALE); // mapped
  gemm_bn_wmma<<<gemmBlocks, 256, 0, stream>>>(x, WZ2, bufC, N, BN_SCALE); // p1
  gemm_bn_wmma<<<gemmBlocks, 256, 0, stream>>>(x, WZ3, bufD, N, BN_SCALE); // p2

  // 2. attention row scores
  s12_kernel<<<rowBlocks, 256, 0, stream>>>(x, bufC, bufD, s1, s2, N);

  // 3. GCN layer 1: y1 = tanh(spmm(a1, h1) + b1)
  fill_f32<<<ndBlocks, 256, 0, stream>>>(bufG, 0.f, (int)nd);
  spmm_scatter<<<e1Waves, 256, 0, stream>>>(a1r, a1c, a1v, bufA, bufG, E1);
  bias_tanh<<<ndBlocks, 256, 0, stream>>>(bufG, b_gcn1, (int)nd);

  // 4. edge softmax on a2
  fill_u32<<<nBlocks, 256, 0, stream>>>(mxu, ENC_NEG_INF, N);
  fill_f32<<<nBlocks, 256, 0, stream>>>(denom, 0.f, N);
  edge_score<<<e2Blocks, 256, 0, stream>>>(a2r, a2c, a2v, s1, s2, edgeE, mxu, E2);
  edge_exp<<<e2Blocks, 256, 0, stream>>>(a2r, edgeE, mxu, denom, E2);

  // 5. y2 = segment_sum(a * mapped[cols])
  fill_f32<<<ndBlocks, 256, 0, stream>>>(bufC, 0.f, (int)nd);
  att_scatter<<<e2Waves, 256, 0, stream>>>(a2r, a2c, edgeE, denom, bufB, bufC, E2);

  // 6. highway: gate = relu(tanh(bn(y2) @ w_hw)); y3
  gemm_bn_wmma<<<gemmBlocks, 256, 0, stream>>>(bufC, WZ4, bufD, N, BN_SCALE);
  highway<<<ndBlocks, 256, 0, stream>>>(bufG, bufC, bufD, bufA, (int)nd); // -> y3 in A

  // 7. GCN layer 2: y4 = tanh(spmm(a1, bn(y3) @ w_gcn2) + b2)
  gemm_bn_wmma<<<gemmBlocks, 256, 0, stream>>>(bufA, WZ5, bufB, N, BN_SCALE);
  fill_f32<<<ndBlocks, 256, 0, stream>>>(bufD, 0.f, (int)nd);
  spmm_scatter<<<e1Waves, 256, 0, stream>>>(a1r, a1c, a1v, bufB, bufD, E1);
  bias_tanh<<<ndBlocks, 256, 0, stream>>>(bufD, b_gcn2, (int)nd);

  // 8. final normalized concat
  final_norm<<<rowBlocks, 256, 0, stream>>>(bufA, bufD, x, (float*)d_out, N);
}
